// TripletLoss_86157043958222
// MI455X (gfx1250) — matrix operand order repaired and checked by reading, not compiled, not run on hardware
//
#include <hip/hip_runtime.h>
#include <hip/hip_bf16.h>

// ---------------------------------------------------------------------------
// Fused triplet-loss pipeline for MI455X (gfx1250, wave32, WMMA + TDM).
//   1) rows_kernel  : sq[i]=||xs_i||^2, hist[label]++, and bf16 hi/lo split of
//                     xs into workspace (so the GEMM loop does zero cvt VALU)
//   2) gram_kernel  : symmetric tiled bf16x3 WMMA Gram. B panels are streamed
//                     into double-buffered LDS by the Tensor Data Mover
//                     (tensor_load_to_lds + s_wait_tensorcnt), overlapped with
//                     WMMA compute. Off-diagonal 128x128 tiles feed BOTH row
//                     sums and (transposed) column sums -> 2x less WMMA work.
//   3) finish_kernel: per-row ap/an means -> margin loss scalar
// ---------------------------------------------------------------------------

typedef __attribute__((ext_vector_type(16))) __bf16       v16bf;
typedef __attribute__((ext_vector_type(8)))  __bf16       v8bf;
typedef __attribute__((ext_vector_type(8)))  float        v8f;
typedef __attribute__((ext_vector_type(4)))  unsigned int u32x4;
typedef __attribute__((ext_vector_type(8)))  int          i32x8;
typedef __attribute__((ext_vector_type(4)))  int          i32x4;

#define NPTS   8192
#define DIM    256
#define BM     128     // rows per workgroup (8 waves x 16)
#define BN     128     // cols per workgroup (8 x 16 per wave)
#define KC     32      // K chunk (bf16 WMMA depth)
#define MARGIN 0.5f
#define EPSV   1e-10f

#if defined(__has_builtin)
#if __has_builtin(__builtin_amdgcn_tensor_load_to_lds) && \
    __has_builtin(__builtin_amdgcn_s_wait_tensorcnt)
#define USE_TDM 1
#endif
#endif
#ifndef USE_TDM
#define USE_TDM 0
#endif

__device__ __forceinline__ float sumsq4(float4 a) {
    return a.x * a.x + a.y * a.y + a.z * a.z + a.w * a.w;
}

// -------------------- 1) row norms + histogram + bf16 split ----------------
__global__ __launch_bounds__(256) void rows_kernel(const float* __restrict__ xs,
                                                   const int* __restrict__ ys,
                                                   float* __restrict__ sq,
                                                   int* __restrict__ hist,
                                                   __bf16* __restrict__ xhi,
                                                   __bf16* __restrict__ xlo) {
    int wave = threadIdx.x >> 5;
    int lane = threadIdx.x & 31;
    int row  = blockIdx.x * 8 + wave;               // 1024 blocks x 8 waves
    size_t base = (size_t)row * DIM + lane * 8;
    float4 a = *(const float4*)(xs + base);
    float4 b = *(const float4*)(xs + base + 4);
    float v[8] = {a.x, a.y, a.z, a.w, b.x, b.y, b.z, b.w};
    v8bf h, l;
#pragma unroll
    for (int j = 0; j < 8; ++j) {
        float  x  = v[j];
        __bf16 hh = (__bf16)x;
        h[j] = hh;
        l[j] = (__bf16)(x - (float)hh);
    }
    *(v8bf*)&xhi[base] = h;
    *(v8bf*)&xlo[base] = l;

    float s = sumsq4(a) + sumsq4(b);
#pragma unroll
    for (int m = 1; m < 32; m <<= 1) s += __shfl_xor(s, m, 32);
    if (lane == 0) {
        sq[row] = s;
        atomicAdd(&hist[ys[row]], 1);
    }
}

// -------------------- TDM descriptor: 2-D bf16 tile (128 rows x 32 elems) --
#if USE_TDM
__device__ __forceinline__ void tdm_load_panel(unsigned int lds_byte_addr,
                                               unsigned long long gaddr) {
    // D# group 0: count=1 | lds_addr | global_addr[56:0] | type=2
    u32x4 g0;
    g0.x = 1u;                                            // count=1, user mode
    g0.y = lds_byte_addr;                                 // LDS dest (bytes)
    g0.z = (unsigned int)(gaddr & 0xFFFFFFFFull);         // global_addr[31:0]
    g0.w = (unsigned int)((gaddr >> 32) & 0x01FFFFFFull)  // global_addr[56:32]
         | (2u << 30);                                    // type = "image"
    // D# group 1: data_size=2B, tensor 256 x 8192, tile 32 x 128, stride 256
    i32x8 g1;
    g1[0] = 0x00010000;      // workgroup_mask=0 | data_size=1 (2 bytes)
    g1[1] = 0x01000000;      // tensor_dim0 = 256  (bits [79:48], low16 here)
    g1[2] = 0x20000000;      // tensor_dim1 = 8192 (bits [111:80], low16 here)
    g1[3] = 0x00200000;      // tile_dim0 = 32     (bits [127:112])
    g1[4] = 128;             // tile_dim1 = 128    (bits [143:128])
    g1[5] = 256;             // tensor_dim0_stride = 256 (bits [207:160])
    g1[6] = 0;
    g1[7] = 0;
    i32x4 z4 = {0, 0, 0, 0};                 // groups 2/3 unused for 2-D tiles
    i32x8 z8 = {0, 0, 0, 0, 0, 0, 0, 0};     // extra group (clang-23 form)
    __builtin_amdgcn_tensor_load_to_lds(g0, g1, z4, z4, z8, 0);
}
#endif

// -------------------- 2) fused symmetric WMMA Gram + distance reduce -------
__global__ __launch_bounds__(256) void gram_kernel(const __bf16* __restrict__ xhi,
                                                   const __bf16* __restrict__ xlo,
                                                   const int* __restrict__ ys,
                                                   const float* __restrict__ sq,
                                                   float* __restrict__ apsum,
                                                   float* __restrict__ totsum) {
    const int bi = blockIdx.x;           // row block
    const int bj = blockIdx.y;           // col block
    if (bi > bj) return;                 // upper triangle only (uniform exit)
    const bool diag = (bi == bj);

    // Double-buffered B panel (128 cols x 32 K, bf16 hi/lo): 2 x 16 KB
    __shared__ alignas(64) __bf16 Bhi[2][BN * KC];
    __shared__ alignas(64) __bf16 Blo[2][BN * KC];
    __shared__ float colap_s[BN];
    __shared__ float coltot_s[BN];

    const int tid  = threadIdx.x;
    const int wave = tid >> 5;
    const int lane = tid & 31;
    const int hi16 = lane >> 4;          // 0: lanes 0-15, 1: lanes 16-31
    const int l15  = lane & 15;

    const int rowBase = bi * BM + wave * 16;
    const int colBase = bj * BN;

    if (tid < BN) { colap_s[tid] = 0.0f; coltot_s[tid] = 0.0f; }

    v8f acc[8] = {v8f{}, v8f{}, v8f{}, v8f{}, v8f{}, v8f{}, v8f{}, v8f{}};

    // A-fragment rows for this lane (contiguous k order, same as B:
    // lanes 0-15 row M=lane hold K 0..15; lanes 16-31 hold K 16..31)
    const __bf16* ahrow = xhi + (size_t)(rowBase + l15) * DIM;
    const __bf16* alrow = xlo + (size_t)(rowBase + l15) * DIM;

    const size_t panelElem = (size_t)colBase * DIM;   // [col][k] row-major

#if USE_TDM
    // Prime buffer 0 with K chunk 0 (TDM ignores EXEC; one wave issues)
    if (wave == 0) {
        tdm_load_panel((unsigned int)(uintptr_t)(void*)&Bhi[0][0],
                       (unsigned long long)(uintptr_t)(xhi + panelElem + 0));
        tdm_load_panel((unsigned int)(uintptr_t)(void*)&Blo[0][0],
                       (unsigned long long)(uintptr_t)(xlo + panelElem + 0));
        __builtin_amdgcn_s_wait_tensorcnt(0);
    }
    __syncthreads();
#endif

    int p = 0;
    for (int it = 0; it < DIM / KC; ++it, p ^= 1) {
        const int k0 = it * KC;
#if USE_TDM
        // Prefetch next K chunk into the other buffer while we compute
        if (wave == 0 && it + 1 < DIM / KC) {
            tdm_load_panel((unsigned int)(uintptr_t)(void*)&Bhi[p ^ 1][0],
                           (unsigned long long)(uintptr_t)(xhi + panelElem + k0 + KC));
            tdm_load_panel((unsigned int)(uintptr_t)(void*)&Blo[p ^ 1][0],
                           (unsigned long long)(uintptr_t)(xlo + panelElem + k0 + KC));
        }
#else
        // Fallback: cooperative bf16 copy (no conversion needed)
        {
            __syncthreads();
            int c  = tid >> 1;            // 0..127
            int kq = (tid & 1) * 16;      // 0 or 16
            size_t src = panelElem + (size_t)c * DIM + k0 + kq;
            *(v8bf*)&Bhi[p][c * KC + kq]     = *(const v8bf*)&xhi[src];
            *(v8bf*)&Bhi[p][c * KC + kq + 8] = *(const v8bf*)&xhi[src + 8];
            *(v8bf*)&Blo[p][c * KC + kq]     = *(const v8bf*)&xlo[src];
            *(v8bf*)&Blo[p][c * KC + kq + 8] = *(const v8bf*)&xlo[src + 8];
            __syncthreads();
        }
#endif
        // A fragments: two 16B bf16 runs per lane, no conversion
        int  kbase = k0 + hi16 * 16;
        v8bf h0 = *(const v8bf*)(ahrow + kbase);
        v8bf h1 = *(const v8bf*)(ahrow + kbase + 8);
        v8bf l0 = *(const v8bf*)(alrow + kbase);
        v8bf l1 = *(const v8bf*)(alrow + kbase + 8);
        v16bf ahi = __builtin_shufflevector(h0, h1, 0, 1, 2, 3, 4, 5, 6, 7,
                                            8, 9, 10, 11, 12, 13, 14, 15);
        v16bf alo = __builtin_shufflevector(l0, l1, 0, 1, 2, 3, 4, 5, 6, 7,
                                            8, 9, 10, 11, 12, 13, 14, 15);

        // B fragment K-offset (contiguous k, as A: lanes 0-15 K=0..15, 16-31 K=16..31)
        const int     boff = hi16 * 16;
        const __bf16* bhp  = &Bhi[p][0];
        const __bf16* blp  = &Blo[p][0];
#pragma unroll
        for (int t = 0; t < 8; ++t) {
            int   c   = t * 16 + l15;
            v16bf bhi = *(const v16bf*)&bhp[c * KC + boff];
            v16bf blo = *(const v16bf*)&blp[c * KC + boff];
            // bf16x3: hi*hi + hi*lo + lo*hi  (~fp32 accurate)
            acc[t] = __builtin_amdgcn_wmma_f32_16x16x32_bf16(
                false, ahi, false, bhi, (short)0, acc[t], false, false);
            acc[t] = __builtin_amdgcn_wmma_f32_16x16x32_bf16(
                false, ahi, false, blo, (short)0, acc[t], false, false);
            acc[t] = __builtin_amdgcn_wmma_f32_16x16x32_bf16(
                false, alo, false, bhi, (short)0, acc[t], false, false);
        }
#if USE_TDM
        if (wave == 0) __builtin_amdgcn_s_wait_tensorcnt(0);
        __syncthreads();   // next-iter panel landed; this-iter panel now free
#endif
    }

    // --- Epilogue: C layout = VGPR v -> M = v + 8*hi16, N = l15 per 16-col tile
    float sr[8];
    int   yr[8];
#pragma unroll
    for (int v = 0; v < 8; ++v) {
        int row = rowBase + v + hi16 * 8;
        sr[v] = sq[row];
        yr[v] = ys[row];
    }

    float rowap[8]  = {0, 0, 0, 0, 0, 0, 0, 0};
    float rowtot[8] = {0, 0, 0, 0, 0, 0, 0, 0};

#pragma unroll
    for (int t = 0; t < 8; ++t) {
        int   cidx = t * 16 + l15;
        int   col  = colBase + cidx;
        float sc   = sq[col];
        int   ycv  = ys[col];
        float cap = 0.0f, ctot = 0.0f;
#pragma unroll
        for (int v = 0; v < 8; ++v) {
            float d2   = sr[v] + sc - 2.0f * acc[t][v];
            float dist = sqrtf(fmaxf(d2, EPSV));
            rowtot[v] += dist;
            ctot      += dist;
            if (yr[v] == ycv) { rowap[v] += dist; cap += dist; }
        }
        if (!diag) {
            // fold the two lane-halves (rows v and v+8 share this column)
            cap  += __shfl_xor(cap, 16, 32);
            ctot += __shfl_xor(ctot, 16, 32);
            if (hi16 == 0) {
                atomicAdd(&colap_s[cidx], cap);     // ds_add_f32
                atomicAdd(&coltot_s[cidx], ctot);
            }
        }
    }

    // Row sums: reduce across the 16 lanes sharing each row (masks < 16)
#pragma unroll
    for (int v = 0; v < 8; ++v) {
        float ap = rowap[v], tot = rowtot[v];
#pragma unroll
        for (int m = 1; m < 16; m <<= 1) {
            ap  += __shfl_xor(ap, m, 32);
            tot += __shfl_xor(tot, m, 32);
        }
        if (l15 == 0) {
            int row = rowBase + v + hi16 * 8;
            atomicAdd(&apsum[row], ap);
            atomicAdd(&totsum[row], tot);
        }
    }

    // Column sums (transposed tile) -> one global atomic per column per block
    if (!diag) {
        __syncthreads();
        if (tid < BN) {
            int col = colBase + tid;
            atomicAdd(&apsum[col], colap_s[tid]);
            atomicAdd(&totsum[col], coltot_s[tid]);
        }
    }
}

// -------------------- 3) final loss ----------------------------------------
__global__ __launch_bounds__(256) void finish_kernel(const int* __restrict__ ys,
                                                     const float* __restrict__ apsum,
                                                     const float* __restrict__ totsum,
                                                     const int* __restrict__ hist,
                                                     float* __restrict__ out) {
    __shared__ float red[256];
    float s = 0.0f;
    for (int r = threadIdx.x; r < NPTS; r += 256) {
        float pc = (float)hist[ys[r]];
        float nc = (float)NPTS - pc;
        float a  = apsum[r];
        float ap = a / pc;
        float an = (totsum[r] - a) / nc;
        s += fmaxf(ap - an + MARGIN, 0.0f);
    }
    red[threadIdx.x] = s;
    __syncthreads();
    for (int off = 128; off > 0; off >>= 1) {
        if (threadIdx.x < off) red[threadIdx.x] += red[threadIdx.x + off];
        __syncthreads();
    }
    if (threadIdx.x == 0) out[0] = red[0] / (float)NPTS;
}

// ---------------------------------------------------------------------------
extern "C" void kernel_launch(void* const* d_in, const int* in_sizes, int n_in,
                              void* d_out, int out_size, void* d_ws, size_t ws_size,
                              hipStream_t stream) {
    const float* xs = (const float*)d_in[0];
    const int*   ys = (const int*)d_in[1];

    // workspace: xhi[8192*256] bf16 | xlo[...] | sq | apsum | totsum | hist
    // total ~ 8.1 MB
    __bf16* xhi    = (__bf16*)d_ws;
    __bf16* xlo    = xhi + (size_t)NPTS * DIM;
    float*  sq     = (float*)(xlo + (size_t)NPTS * DIM);
    float*  apsum  = sq + NPTS;
    float*  totsum = apsum + NPTS;
    int*    hist   = (int*)(totsum + NPTS);

    (void)hipMemsetAsync(apsum, 0, (size_t)(2 * NPTS + 64) * sizeof(float), stream);

    rows_kernel<<<NPTS / 8, 256, 0, stream>>>(xs, ys, sq, hist, xhi, xlo);

    dim3 grid(NPTS / BM, NPTS / BN);   // 64 x 64, upper triangle does work
    gram_kernel<<<grid, 256, 0, stream>>>(xhi, xlo, ys, sq, apsum, totsum);

    finish_kernel<<<1, 256, 0, stream>>>(ys, apsum, totsum, hist, (float*)d_out);
}
